// KDALayer_16415365005651
// MI455X (gfx1250) — matrix operand order repaired and checked by reading, not compiled, not run on hardware
//
#include <hip/hip_runtime.h>
#include <hip/hip_bf16.h>
#include <math.h>

// ---------------- problem constants (match reference) ----------------
#define DIMC      2048
#define HC        16
#define KC        128
#define VC        128
#define BC        2
#define TC        2048
#define BT        (BC * TC)            // 4096 rows
#define QOFF      0
#define KOFF      (HC * KC)            // 2048
#define VOFF      (2 * HC * KC)        // 4096
#define GOFF      (VOFF + HC * VC)     // 6144
#define BETAOFF   (GOFF + HC * KC)     // 8192
#define TOTAL_OUT (BETAOFF + HC)       // 8208
#define HV        (HC * VC)            // 2048

// ---------------- WMMA vector types (CDNA5 wave32) ----------------
typedef __attribute__((ext_vector_type(16))) __bf16 v16bf;
typedef __attribute__((ext_vector_type(8)))  float  v8f;

// fp32 -> bf16 round-to-nearest-even, and back
static __device__ __forceinline__ unsigned short f2bf(float f) {
  unsigned u = __builtin_bit_cast(unsigned, f);
  u += 0x7fffu + ((u >> 16) & 1u);
  return (unsigned short)(u >> 16);
}
static __device__ __forceinline__ float bf2f(unsigned short h) {
  unsigned u = ((unsigned)h) << 16;
  return __builtin_bit_cast(float, u);
}
static __device__ __forceinline__ __bf16 us2bf(unsigned short h) {
  return __builtin_bit_cast(__bf16, h);
}

// ---- gfx1250 async global->LDS copy (16B per lane), tracked by ASYNCcnt ----
// LDS byte offset is the low 32 bits of the generic pointer to a __shared__
// object (flat LDS aperture truncation, ISA 10.2).
static __device__ __forceinline__ void async_cp16(const float* __restrict__ g,
                                                  float* l) {
  unsigned loff = (unsigned)(uintptr_t)l;
  asm volatile("global_load_async_to_lds_b128 %0, %1, off"
               :: "v"(loff), "v"(g) : "memory");
}
#define WAIT_ASYNC(n) asm volatile("s_wait_asynccnt %0" :: "i"(n) : "memory")

// ---------------- split-bf16 GEMM: C[M][N] = A[M][Kd] * B[Kd][N] ----------------
// 256 threads = 8 waves; block tile 128x128; BK=32; double-buffered async LDS.
// Wave grid 4(m) x 2(n); each wave owns a 32x64 patch = 2x4 tiles of 16x16,
// 3 WMMAs per tile per K-step -> 24 static v_wmma per K iteration.
#define BM 128
#define BN 128
#define BK 32
#define APAD 4   // A row stride 36 floats = 144B (16B aligned, bank-friendly)
#define BPAD 4   // B row stride 132 floats = 528B (16B aligned)

__global__ __launch_bounds__(256) void kda_gemm_bf16x3(
    const float* __restrict__ A, const float* __restrict__ Bm,
    float* __restrict__ C, int M, int N, int Kd)
{
  __shared__ __align__(16) float As[2][BM][BK + APAD];
  __shared__ __align__(16) float Bs[2][BK][BN + BPAD];

  const int tid  = threadIdx.x;
  const int lane = tid & 31;
  const int wid  = tid >> 5;
  const int wm   = wid & 3;     // wave row (0..3) -> m offset wm*32
  const int wn   = wid >> 2;    // wave col (0..1) -> n offset wn*64
  const int mblk = blockIdx.y * BM;
  const int nblk = blockIdx.x * BN;
  const int l15  = lane & 15;
  const int khalf = lane >> 4;  // 0 or 1

  // Issue one K-tile's async copies: exactly 8 async instructions per wave
  // (4 for the 128x32 A tile, 4 for the 32x128 B tile). B columns are clamped
  // to N-4 (duplicate loads at the N edge; those columns are never stored).
  auto issue = [&](int k0, int buf) {
    #pragma unroll
    for (int i = 0; i < 4; ++i) {
      int idx = tid + i * 256;          // 0..1023
      int r   = idx >> 3;               // row 0..127
      int c4  = (idx & 7) << 2;         // col 0,4,..,28
      async_cp16(&A[(size_t)(mblk + r) * Kd + k0 + c4], &As[buf][r][c4]);
    }
    #pragma unroll
    for (int i = 0; i < 4; ++i) {
      int idx = tid + i * 256;          // 0..1023
      int r   = idx >> 5;               // row 0..31
      int c4  = (idx & 31) << 2;        // col 0,4,..,124
      int cc  = nblk + c4;
      if (cc > N - 4) cc = N - 4;       // N is a multiple of 4 here
      async_cp16(&Bm[(size_t)(k0 + r) * N + cc], &Bs[buf][r][c4]);
    }
  };

  v8f acc[2][4] = {};
  const int nk = Kd / BK;

  issue(0, 0);
  for (int it = 0; it < nk; ++it) {
    const int buf = it & 1;
    if (it + 1 < nk) {
      issue((it + 1) * BK, buf ^ 1);    // overlap next tile with this compute
      WAIT_ASYNC(8);                    // <=8 outstanding -> tile `it` landed
    } else {
      WAIT_ASYNC(0);
    }
    __syncthreads();                    // all waves' copies visible

    const float (*Asb)[BK + APAD] = As[buf];
    const float (*Bsb)[BN + BPAD] = Bs[buf];

    // ---- build hi/lo bf16 fragments per ISA 7.12.2 16-bit layouts ----
    // A 16x32: lane<16 -> M=lane, elems 0..7 = K0..7, 8..15 = K16..23
    //          lane>=16 -> elems 0..7 = K8..15, 8..15 = K24..31
    v16bf afh[2], afl[2];
    #pragma unroll
    for (int sm = 0; sm < 2; ++sm) {
      const int row = wm * 32 + sm * 16 + l15;
      const float* ap = &Asb[row][khalf * 8];
      #pragma unroll
      for (int e = 0; e < 8; ++e) {
        float x0 = ap[e];               // K = khalf*8 + e
        float x1 = ap[16 + e];          // K = 16 + khalf*8 + e
        unsigned short h0 = f2bf(x0), h1 = f2bf(x1);
        afh[sm][e]     = us2bf(h0);
        afh[sm][e + 8] = us2bf(h1);
        afl[sm][e]     = us2bf(f2bf(x0 - bf2f(h0)));
        afl[sm][e + 8] = us2bf(f2bf(x1 - bf2f(h1)));
      }
    }
    // B 32x16: lane = column; elems 0..7 = K khalf*8+0..7, 8..15 = K 16+khalf*8+0..7
    v16bf bfh[4], bfl[4];
    #pragma unroll
    for (int sn = 0; sn < 4; ++sn) {
      const int col = wn * 64 + sn * 16 + l15;
      #pragma unroll
      for (int e = 0; e < 16; ++e) {
        float x = Bsb[khalf * 8 + e + ((e >> 3) << 3)][col];
        unsigned short h = f2bf(x);
        bfh[sn][e] = us2bf(h);
        bfl[sn][e] = us2bf(f2bf(x - bf2f(h)));
      }
    }

    // ---- 3-term split-bf16 accumulation (drop lo*lo): ~fp32 accuracy ----
    #pragma unroll
    for (int sm = 0; sm < 2; ++sm) {
      #pragma unroll
      for (int sn = 0; sn < 4; ++sn) {
        v8f c = acc[sm][sn];
        c = __builtin_amdgcn_wmma_f32_16x16x32_bf16(false, afh[sm], false, bfh[sn],
                                                    (short)0, c, false, false);
        c = __builtin_amdgcn_wmma_f32_16x16x32_bf16(false, afh[sm], false, bfl[sn],
                                                    (short)0, c, false, false);
        c = __builtin_amdgcn_wmma_f32_16x16x32_bf16(false, afl[sm], false, bfh[sn],
                                                    (short)0, c, false, false);
        acc[sm][sn] = c;
      }
    }
    __syncthreads();                    // done reading buf before it refills
  }

  // ---- store D: VGPR r -> row m0+r (lanes<16) / m0+r+8 (lanes>=16)
  const int rbump = (lane >= 16) ? 8 : 0;
  #pragma unroll
  for (int sm = 0; sm < 2; ++sm) {
    #pragma unroll
    for (int sn = 0; sn < 4; ++sn) {
      int col = nblk + wn * 64 + sn * 16 + l15;
      if (col < N) {
        #pragma unroll
        for (int r = 0; r < 8; ++r) {
          int row = mblk + wm * 32 + sm * 16 + r + rbump;
          C[(size_t)row * N + col] = acc[sm][sn][r];
        }
      }
    }
  }
}

// ---------------- KDA gated delta-rule scan ----------------
// One workgroup per (b,h). 128 threads; thread v keeps state column S[:,v]
// (128 f32) in VGPRs. q/k/exp(g) staged in LDS per step, broadcast-read in
// fully unrolled FMA loops.
__global__ __launch_bounds__(128) void kda_scan(
    const float* __restrict__ proj, float* __restrict__ y)
{
  const int bh = blockIdx.x;     // 0..31
  const int b  = bh >> 4;        // H = 16
  const int h  = bh & 15;
  const int v  = threadIdx.x;    // 0..127

  __shared__ float qs[KC];
  __shared__ float ks[KC];
  __shared__ float eg[KC];
  __shared__ float betas;

  float S[KC];
  #pragma unroll
  for (int k = 0; k < KC; ++k) S[k] = 0.f;

  const float scale = 0.08838834764831845f;  // 128^-0.5

  for (int t = 0; t < TC; ++t) {
    const size_t base = (size_t)(b * TC + t) * TOTAL_OUT;
    const float qv = proj[base + QOFF + h * KC + v];
    const float kv = proj[base + KOFF + h * KC + v];
    const float vv = proj[base + VOFF + h * VC + v];
    const float gr = proj[base + GOFF + h * KC + v];
    // g = -0.1 * softplus(g_raw); eg = exp(g)
    const float sp = (gr > 20.f) ? gr : log1pf(__expf(gr));
    qs[v] = qv * scale;
    ks[v] = kv;
    eg[v] = __expf(-0.1f * sp);
    if (v == 0) {
      const float br = proj[base + BETAOFF + h];
      betas = 1.f / (1.f + __expf(-br));
    }
    __syncthreads();

    // err_v = v_t[v] - sum_k k[k] * (S[k][v]*eg[k]);  S <- decayed state
    float err = vv;
    #pragma unroll
    for (int k = 0; k < KC; ++k) {
      const float sd = S[k] * eg[k];
      err = fmaf(-ks[k], sd, err);
      S[k] = sd;
    }
    // S[k][v] += beta * k[k] * err_v;  y_v = sum_k q[k] * S[k][v]
    const float be = betas * err;
    float yv = 0.f;
    #pragma unroll
    for (int k = 0; k < KC; ++k) {
      S[k] = fmaf(ks[k], be, S[k]);
      yv = fmaf(qs[k], S[k], yv);
    }
    y[(size_t)(b * TC + t) * HV + h * VC + v] = yv;
    __syncthreads();
  }
}

// ---------------- launch ----------------
extern "C" void kernel_launch(void* const* d_in, const int* in_sizes, int n_in,
                              void* d_out, int out_size, void* d_ws, size_t ws_size,
                              hipStream_t stream) {
  (void)in_sizes; (void)n_in; (void)out_size; (void)ws_size;
  const float* x     = (const float*)d_in[0];   // [B,T,DIM] f32
  const float* W_in  = (const float*)d_in[1];   // [DIM, TOTAL_OUT] f32
  const float* W_out = (const float*)d_in[2];   // [HV, DIM] f32
  float* out  = (float*)d_out;                  // [B,T,DIM] f32

  float* proj = (float*)d_ws;                          // BT x TOTAL_OUT
  float* yb   = proj + (size_t)BT * TOTAL_OUT;         // BT x HV

  dim3 blk(256);
  // proj = x @ W_in   (M=4096, N=8208, K=2048)
  dim3 g1((TOTAL_OUT + BN - 1) / BN, BT / BM);
  kda_gemm_bf16x3<<<g1, blk, 0, stream>>>(x, W_in, proj, BT, TOTAL_OUT, DIMC);

  // sequential gated delta-rule scan over T, one WGP-resident block per (b,h)
  kda_scan<<<dim3(BC * HC), dim3(128), 0, stream>>>(proj, yb);

  // out = y @ W_out   (M=4096, N=2048, K=2048)
  dim3 g2((DIMC + BN - 1) / BN, BT / BM);
  kda_gemm_bf16x3<<<g2, blk, 0, stream>>>(yb, W_out, out, BT, DIMC, HV);
}
